// DG_62895501083322
// MI455X (gfx1250) — matrix operand order repaired and checked by reading, not compiled, not run on hardware
//
#include <hip/hip_runtime.h>
#include <stdint.h>

#define B_SZ 512
#define I_SZ 4096
#define H_SZ 8192
#define K_SEL 64
#define DECAY 0.95f

typedef __attribute__((ext_vector_type(16))) __bf16 v16bf;
typedef __attribute__((ext_vector_type(8)))  float  v8f;

union FragBF {
    v16bf    bf;
    unsigned u[8];
};

// bf16(hi) of two fp32 packed into one dword via a single v_perm_b32.
__device__ __forceinline__ unsigned pack_hi(float x0, float x1) {
    return __builtin_amdgcn_perm(__float_as_uint(x1), __float_as_uint(x0), 0x07060302u);
}
// bf16 residuals lo = x - hi, packed the same way.
__device__ __forceinline__ unsigned pack_lo(float x0, float x1) {
    unsigned u0 = __float_as_uint(x0), u1 = __float_as_uint(x1);
    float l0 = x0 - __uint_as_float(u0 & 0xFFFF0000u);
    float l1 = x1 - __uint_as_float(u1 & 0xFFFF0000u);
    return __builtin_amdgcn_perm(__float_as_uint(l1), __float_as_uint(l0), 0x07060302u);
}

// Build hi/lo dwords for 4 consecutive fp32 (two packed pairs).
__device__ __forceinline__ void pack4(const float4 q, unsigned* hi, unsigned* lo) {
    hi[0] = pack_hi(q.x, q.y);
    hi[1] = pack_hi(q.z, q.w);
    lo[0] = pack_lo(q.x, q.y);
    lo[1] = pack_lo(q.z, q.w);
}

// ---------------------------------------------------------------------------
// GEMM: enc[b,h] = sum_i A[b,i] * W[h,i], fp32 in/out, bf16 hi/lo split WMMA.
// Block = 256 threads (8 waves). Block tile 16(M) x 256(N); wave tile 16 x 32
// (two 16x16 N sub-tiles -> 6 WMMAs per K-chunk, A fragments amortized).
// Grid = (H/256, B/16) = (32, 32). K loop in chunks of 32.
// A tile staged global->LDS with CDNA5 async-to-LDS DMA (no VGPR round-trip).
// ---------------------------------------------------------------------------
__global__ __launch_bounds__(256) void dg_gemm_wmma(const float* __restrict__ A,
                                                    const float* __restrict__ W,
                                                    float* __restrict__ C) {
    __shared__ float s_a[16 * 32];   // A tile, row-major [m][k]

    const int lane = threadIdx.x & 31;
    const int wv   = threadIdx.x >> 5;        // 0..7 -> 32-wide N sub-tile
    const int m0   = blockIdx.y * 16;
    const int h0   = blockIdx.x * 256;
    const int nCol = lane & 15;
    const int half = lane >> 4;               // K half-wave selector

    // Async A staging: thread t moves A[m0 + t/16, kc + 2*(t%16) .. +1] -> LDS.
    const unsigned ldsDst =
        (unsigned)(uintptr_t)(&s_a[0]) + (unsigned)threadIdx.x * 8u;
    const float* agp =
        A + (size_t)(m0 + (threadIdx.x >> 4)) * I_SZ + ((threadIdx.x & 15) << 1);

    // B fragments (32x16 bf16): lane (half,N) reads 16 contiguous fp32 of a W row.
    const float* wrow0 = W + (size_t)(h0 + wv * 32 + nCol) * I_SZ + half * 16;
    const float* wrow1 = wrow0 + (size_t)16 * I_SZ;

    v8f acc0 = {}, acc1 = {};

    for (int kc = 0; kc < I_SZ; kc += 32) {
        __syncthreads();   // previous-iteration fragment reads done
        asm volatile("global_load_async_to_lds_b64 %0, %1, off"
                     :: "v"(ldsDst), "v"(agp + kc) : "memory");
        asm volatile("s_wait_asynccnt 0x0" ::: "memory");
        __syncthreads();

        // A fragment (16x32 bf16): lane -> K in [16h,16h+16)
        const float4* arv = (const float4*)&s_a[(lane & 15) * 32 + half * 16];
        FragBF a_hi, a_lo;
        pack4(arv[0], &a_hi.u[0], &a_lo.u[0]);
        pack4(arv[1], &a_hi.u[2], &a_lo.u[2]);
        pack4(arv[2], &a_hi.u[4], &a_lo.u[4]);
        pack4(arv[3], &a_hi.u[6], &a_lo.u[6]);

        const float4* wp0 = (const float4*)(wrow0 + kc);
        const float4* wp1 = (const float4*)(wrow1 + kc);
        FragBF b0_hi, b0_lo, b1_hi, b1_lo;
#pragma unroll
        for (int v = 0; v < 4; v++) {
            pack4(wp0[v], &b0_hi.u[2 * v], &b0_lo.u[2 * v]);
            pack4(wp1[v], &b1_hi.u[2 * v], &b1_lo.u[2 * v]);
        }
        __builtin_prefetch((const void*)(wp0 + 8), 0, 1);   // next K chunk of W
        __builtin_prefetch((const void*)(wp1 + 8), 0, 1);

        // fp32 ~= hi*hi + hi*lo + lo*hi (drop lo*lo), f32 accumulate
        acc0 = __builtin_amdgcn_wmma_f32_16x16x32_bf16(false, a_hi.bf, false, b0_hi.bf,
                                                       (short)0, acc0, false, false);
        acc0 = __builtin_amdgcn_wmma_f32_16x16x32_bf16(false, a_hi.bf, false, b0_lo.bf,
                                                       (short)0, acc0, false, false);
        acc0 = __builtin_amdgcn_wmma_f32_16x16x32_bf16(false, a_lo.bf, false, b0_hi.bf,
                                                       (short)0, acc0, false, false);
        acc1 = __builtin_amdgcn_wmma_f32_16x16x32_bf16(false, a_hi.bf, false, b1_hi.bf,
                                                       (short)0, acc1, false, false);
        acc1 = __builtin_amdgcn_wmma_f32_16x16x32_bf16(false, a_hi.bf, false, b1_lo.bf,
                                                       (short)0, acc1, false, false);
        acc1 = __builtin_amdgcn_wmma_f32_16x16x32_bf16(false, a_lo.bf, false, b1_hi.bf,
                                                       (short)0, acc1, false, false);
    }

    // C/D 16x16 f32 layout: lane = N (mod 16), VGPR r -> M = 8*half + r
    const int outCol0 = h0 + wv * 32 + nCol;
#pragma unroll
    for (int r = 0; r < 8; r++) {
        C[(size_t)(m0 + half * 8 + r) * H_SZ + outCol0]      = acc0[r];
        C[(size_t)(m0 + half * 8 + r) * H_SZ + outCol0 + 16] = acc1[r];
    }
}

// ---------------------------------------------------------------------------
// Sequential inhibition scan + fused final top-k mask.
// One block of 1024 threads; thread t owns columns {t, t+1024, ...} (8 each).
// Inhibition state lives in registers (owner-exclusive columns).
// Per-step exact top-64 via 4x8-bit radix select on monotonic float keys.
// ---------------------------------------------------------------------------
__global__ __launch_bounds__(1024) void dg_scan(const float* __restrict__ enc,
                                                float* __restrict__ out) {
    __shared__ unsigned s_key[H_SZ];   // keys, later reused as zero-flags
    __shared__ unsigned s_hist[256];
    __shared__ unsigned s_prefix, s_remaining, s_eq, s_bound, s_pos, s_bound2;

    const int tid  = threadIdx.x;
    const int lane = tid & 31;

    float inh[8];
#pragma unroll
    for (int j = 0; j < 8; j++) inh[j] = 0.0f;

    for (int b = 0; b < B_SZ; b++) {
        float ev[8];
        unsigned kv[8];
#pragma unroll
        for (int j = 0; j < 8; j++) {
            int idx = tid + j * 1024;
            float e = enc[(size_t)b * H_SZ + idx];
            ev[j] = e;
            float r = fabsf(e) * (1.0f - inh[j]);
            unsigned u   = __float_as_uint(r);
            unsigned key = (u & 0x80000000u) ? ~u : (u | 0x80000000u);
            kv[j] = key;
            s_key[idx] = key;
        }
        if (tid == 0) { s_prefix = 0u; s_remaining = K_SEL; }

        // ---- 4-pass radix select (descending), MSB first ----
#pragma unroll 1
        for (int shift = 24; shift >= 0; shift -= 8) {
            if (tid < 256) s_hist[tid] = 0u;
            __syncthreads();
            unsigned dmask = (shift == 24) ? 0u : (0xFFFFFFFFu << (shift + 8));
            unsigned pref  = s_prefix;
#pragma unroll
            for (int j = 0; j < 8; j++)
                if ((kv[j] & dmask) == (pref & dmask))
                    atomicAdd(&s_hist[(kv[j] >> shift) & 255u], 1u);
            __syncthreads();
            if (tid < 32) {   // single-wave suffix scan over 256 buckets
                unsigned c[8], bsum = 0;
#pragma unroll
                for (int q = 0; q < 8; q++) { c[q] = s_hist[lane * 8 + q]; bsum += c[q]; }
                unsigned suff = bsum;
#pragma unroll
                for (int off = 1; off < 32; off <<= 1) {
                    unsigned t = __shfl_down(suff, off);
                    if (lane + off < 32) suff += t;
                }
                unsigned cum = suff - bsum;       // count strictly above this block
                unsigned rem = s_remaining;
#pragma unroll
                for (int q = 7; q >= 0; q--) {
                    unsigned cnt = c[q];
                    if (cnt > 0u && cum < rem && cum + cnt >= rem) {
                        s_prefix    = (pref & dmask) | ((unsigned)(lane * 8 + q) << shift);
                        s_remaining = rem - cum;
                        if (shift == 0) s_eq = cnt;
                    }
                    cum += cnt;
                }
            }
            __syncthreads();
        }

        // ---- deterministic tie handling (rare): first `need` equals by index ----
        if (tid == 0) {
            unsigned need = s_remaining, e = s_eq;
            unsigned bound = H_SZ;
            if (e != need) {
                unsigned cnt = 0;
                for (int i = 0; i < H_SZ; i++)
                    if (s_key[i] == s_prefix) { cnt++; if (cnt == need) { bound = i + 1; break; } }
            }
            s_bound = bound;
            s_pos   = 0u;
        }
        __syncthreads();

        unsigned kth = s_prefix, bound = s_bound;
        int fired_bits = 0, pos_bits = 0, myPos = 0;
#pragma unroll
        for (int j = 0; j < 8; j++) {
            int  idx   = tid + j * 1024;
            bool fired = (kv[j] > kth) || (kv[j] == kth && (unsigned)idx < bound);
            inh[j] = inh[j] * DECAY + (fired ? 1.0f : 0.0f);
            bool pos = fired && (ev[j] > 0.0f);
            myPos      += pos ? 1 : 0;
            fired_bits |= (fired ? 1 : 0) << j;
            pos_bits   |= (pos   ? 1 : 0) << j;
            s_key[idx]  = (!fired || ev[j] == 0.0f) ? 1u : 0u;   // filtered == 0 flag
        }
#pragma unroll
        for (int off = 16; off >= 1; off >>= 1) myPos += __shfl_down(myPos, off);
        if (lane == 0) atomicAdd(&s_pos, (unsigned)myPos);
        __syncthreads();

        unsigned P = s_pos;
        if (tid == 0) {
            if (P < K_SEL) {   // zero-fill bound: within first 128 indices by construction
                unsigned needz = K_SEL - P, cnt = 0;
                int i = 0;
                for (; i < H_SZ; i++) { cnt += s_key[i]; if (cnt == needz) break; }
                s_bound2 = (unsigned)(i + 1);
            } else {
                s_bound2 = 0u;
            }
        }
        __syncthreads();

        unsigned b2 = s_bound2;
#pragma unroll
        for (int j = 0; j < 8; j++) {
            int idx = tid + j * 1024;
            float o;
            if (P >= K_SEL)
                o = ((fired_bits >> j) & 1) ? 1.0f : 0.0f;
            else
                o = ((pos_bits >> j) & 1) ? 1.0f
                  : ((s_key[idx] != 0u && (unsigned)idx < b2) ? 1.0f : 0.0f);
            out[(size_t)b * H_SZ + idx] = o;
        }
        __syncthreads();
    }
}

extern "C" void kernel_launch(void* const* d_in, const int* in_sizes, int n_in,
                              void* d_out, int out_size, void* d_ws, size_t ws_size,
                              hipStream_t stream) {
    (void)in_sizes; (void)n_in; (void)out_size; (void)ws_size;
    const float* inputs = (const float*)d_in[0];   // [B, I] fp32
    const float* weight = (const float*)d_in[1];   // [H, I] fp32
    // d_in[2] = k (int scalar) — fixed to 64 at compile time per reference.

    float* enc = (float*)d_ws;                     // [B, H] fp32 (16 MB)

    dim3 grid(H_SZ / 256, B_SZ / 16);              // (32, 32)
    dg_gemm_wmma<<<grid, 256, 0, stream>>>(inputs, weight, enc);
    dg_scan<<<1, 1024, 0, stream>>>(enc, (float*)d_out);
}
